// CrossAttention_62929860821524
// MI455X (gfx1250) — matrix operand order repaired and checked by reading, not compiled, not run on hardware
//
#include <hip/hip_runtime.h>

typedef __attribute__((ext_vector_type(16))) _Float16 v16h;
typedef __attribute__((ext_vector_type(8)))  _Float16 v8h;
typedef __attribute__((ext_vector_type(8)))  float    v8f;
typedef __attribute__((ext_vector_type(4)))  unsigned int u32x4;
typedef __attribute__((ext_vector_type(8)))  int      i32x8;
typedef __attribute__((ext_vector_type(4)))  int      i32x4;

namespace {
constexpr int kB  = 2;     // batch
constexpr int kS  = 2048;  // Sq == Sk
constexpr int kD  = 1024;  // model dim
constexpr int kH  = 16;    // heads
constexpr int kDh = 64;    // head dim
}

// ---------------- WMMA fragment helpers (wave32, 16x16x32 f16) ----------------
__device__ __forceinline__ v16h fragA_ld(const _Float16* p) {
  v8h lo = *(const v8h*)(p);
  v8h hi = *(const v8h*)(p + 16);
  v16h r;
#pragma unroll
  for (int i = 0; i < 8; ++i) { r[i] = lo[i]; r[i + 8] = hi[i]; }
  return r;
}

__device__ __forceinline__ v16h fragB_ld(const _Float16* p) {
  v8h lo = *(const v8h*)(p);
  v8h hi = *(const v8h*)(p + 16);
  v16h r;
#pragma unroll
  for (int i = 0; i < 8; ++i) { r[i] = lo[i]; r[i + 8] = hi[i]; }
  return r;
}

__device__ __forceinline__ v8f wmma_f16(v16h a, v16h b, v8f c) {
  return __builtin_amdgcn_wmma_f32_16x16x32_f16(false, a, false, b, (short)0, c,
                                                false, false);
}

// ---------------- Tensor Data Mover: 2D tile global -> LDS --------------------
// D# per CDNA5 ISA ch.8: group0 = {count=1, lds_addr, global_addr, type=2};
// group1 = {data_size=2B, pad fields, tensor_dim0/1, tile_dim0/1, dim0_stride}.
// pad_interval code: row bytes = 2^(code+1) DWORDs; pad_amount code: pad =
// (code+1) DWORDs. Issued by one wave; tracked with TENSORcnt.
__device__ __forceinline__ void tdm_load_2d(unsigned lds_addr, const void* gptr,
                                            unsigned tensor_d0,
                                            unsigned tensor_d1,
                                            unsigned stride_d0,
                                            unsigned tile_d0, unsigned tile_d1,
                                            unsigned pad_interval_code,
                                            unsigned pad_amount_code) {
  unsigned long long ga = (unsigned long long)gptr;
  u32x4 g0;
  g0[0] = 1u;                                   // count=1, user descriptor
  g0[1] = lds_addr;                             // LDS byte address
  g0[2] = (unsigned)(ga & 0xffffffffu);         // global_addr[31:0]
  g0[3] = (unsigned)((ga >> 32) & 0x01ffffffu)  // global_addr[56:32]
          | (2u << 30);                         // type = 2 ("image")
  i32x8 g1;
  unsigned w0 = (1u << 16);                     // data_size = 1 -> 2 bytes
  w0 |= (1u << 20) | (pad_interval_code << 22) | (pad_amount_code << 25);
  g1[0] = (int)w0;
  g1[1] = (int)((tensor_d0 & 0xffffu) << 16);   // tensor_dim0[15:0] @ bits 48+
  g1[2] = (int)(((tensor_d0 >> 16) & 0xffffu) | ((tensor_d1 & 0xffffu) << 16));
  g1[3] = (int)(((tensor_d1 >> 16) & 0xffffu) | ((tile_d0 & 0xffffu) << 16));
  g1[4] = (int)(tile_d1 & 0xffffu);             // tile_dim1; tile_dim2 = 0
  g1[5] = (int)stride_d0;                       // tensor_dim0_stride[31:0]
  g1[6] = 0;
  g1[7] = 0;
  i32x4 z4 = {0, 0, 0, 0};
#if __clang_major__ >= 23
  i32x8 z8 = {0, 0, 0, 0, 0, 0, 0, 0};
  __builtin_amdgcn_tensor_load_to_lds(g0, g1, z4, z4, z8, 0);
#else
  __builtin_amdgcn_tensor_load_to_lds(g0, g1, z4, z4, 0);
#endif
}

__device__ __forceinline__ unsigned lds_off(const void* p) {
  return (unsigned)(unsigned long long)(uintptr_t)p;  // LDS offset in low 32b
}

// ---------------- one-shot pre-passes ----------------------------------------
__global__ __launch_bounds__(256) void cvt_f16_kernel(
    const float* __restrict__ x, _Float16* __restrict__ y) {
  size_t i = ((size_t)blockIdx.x * 256 + threadIdx.x) * 8;
  float4 a = *(const float4*)(x + i);
  float4 b = *(const float4*)(x + i + 4);
  v8h o;
  o[0] = (_Float16)a.x; o[1] = (_Float16)a.y;
  o[2] = (_Float16)a.z; o[3] = (_Float16)a.w;
  o[4] = (_Float16)b.x; o[5] = (_Float16)b.y;
  o[6] = (_Float16)b.z; o[7] = (_Float16)b.w;
  *(v8h*)(y + i) = o;
}

// W[1024,1024] f32 row-major [k][n] -> Wt f16 [n][k], LDS-tiled 64x64
__global__ __launch_bounds__(256) void wtrans_kernel(
    const float* __restrict__ W, _Float16* __restrict__ Wt) {
  __shared__ _Float16 t[64 * 72];
  const int tid = threadIdx.x;
  const int k0 = blockIdx.x * 64, n0 = blockIdx.y * 64;
#pragma unroll
  for (int c = 0; c < 4; ++c) {
    int ch = tid + c * 256;
    int kk = ch >> 4, nc = (ch & 15) * 4;
    float4 f = *(const float4*)(W + (size_t)(k0 + kk) * kD + n0 + nc);
    t[kk * 72 + nc + 0] = (_Float16)f.x;
    t[kk * 72 + nc + 1] = (_Float16)f.y;
    t[kk * 72 + nc + 2] = (_Float16)f.z;
    t[kk * 72 + nc + 3] = (_Float16)f.w;
  }
  __syncthreads();
#pragma unroll
  for (int c = 0; c < 2; ++c) {
    int ch = tid + c * 256;
    int nn = ch >> 3, kc = (ch & 7) * 8;
    v8h o;
#pragma unroll
    for (int j = 0; j < 8; ++j) o[j] = t[(kc + j) * 72 + nn];
    *(v8h*)(Wt + (size_t)(n0 + nn) * kD + k0 + kc) = o;
  }
}

// ---------------- f16 GEMM + bias: Y[M,1024] = X @ Wt^T + b ------------------
// Block tile 128x128, BK=32, 8 waves; wave = 2 row-tiles x 4 col-tiles ->
// 8 WMMA / k-step. Tiles staged by the TDM (wave 0 issues, TENSORcnt waits),
// other waves' issue slots stay free for WMMA.
template <bool OUT_HEAD>
__global__ __launch_bounds__(256) void gemm_f16_kernel(
    const _Float16* __restrict__ A, const _Float16* __restrict__ Wt,
    const float* __restrict__ bias, void* __restrict__ Outp, float oscale) {
  constexpr int LDA = 40, LDB = 40;   // 32 + 8 pad halves (TDM pad: 16B)
  __shared__ _Float16 ldsA[128 * LDA];
  __shared__ _Float16 ldsBt[128 * LDB];

  const int tid = threadIdx.x;
  const int m0 = blockIdx.x * 128;
  const int n0 = blockIdx.y * 128;
  const int w = tid >> 5, lane = tid & 31;
  const int l16 = lane & 15, lhi = (lane >> 4) & 1;
  const int rowTile = (w & 3) * 32;
  const int colBase = (w >> 2) * 64;

  v8f acc[2][4] = {};

  for (int kt = 0; kt < kD / 32; ++kt) {
    const int k0 = kt * 32;
    __syncthreads();
    if (tid < 32) {   // wave 0: DMA both tiles. Rows = 64B -> interval code 3.
      tdm_load_2d(lds_off(ldsA), A + (size_t)m0 * kD + k0,
                  kD, kB * kS, kD, 32, 128, 3, 3);
      tdm_load_2d(lds_off(ldsBt), Wt + (size_t)n0 * kD + k0,
                  kD, kD, kD, 32, 128, 3, 3);
      __builtin_amdgcn_s_wait_tensorcnt(0);
    }
    __syncthreads();

    v16h a0 = fragA_ld(&ldsA[(rowTile + l16) * LDA + lhi * 8]);
    v16h a1 = fragA_ld(&ldsA[(rowTile + 16 + l16) * LDA + lhi * 8]);
#pragma unroll
    for (int j = 0; j < 4; ++j) {
      v16h b = fragB_ld(&ldsBt[(colBase + j * 16 + l16) * LDB + lhi * 8]);
      acc[0][j] = wmma_f16(a0, b, acc[0][j]);
      acc[1][j] = wmma_f16(a1, b, acc[1][j]);
    }
  }

#pragma unroll
  for (int t = 0; t < 2; ++t)
#pragma unroll
    for (int j = 0; j < 4; ++j)
#pragma unroll
      for (int r = 0; r < 8; ++r) {
        int row = m0 + rowTile + t * 16 + r + lhi * 8;
        int col = n0 + colBase + j * 16 + l16;
        float val = (acc[t][j][r] + bias[col]) * oscale;
        if (OUT_HEAD) {
          int bb = row / kS, s = row % kS;
          int hh = col >> 6, dh = col & 63;
          ((_Float16*)Outp)[((size_t)(bb * kH + hh) * kS + s) * kDh + dh] =
              (_Float16)val;
        } else {
          ((float*)Outp)[(size_t)row * kD + col] = val;
        }
      }
}

// ---------------- Flash attention, 64-key tiles -------------------------------
// K tile staged by TDM; V tile loaded+transposed by the waves while the TDM
// runs. Per tile: 8 WMMA (Q.K^T) + 2 WMMA (rowsum via ones) + 8 WMMA (P.V).
__global__ __launch_bounds__(256) void flash_attn_kernel(
    const _Float16* __restrict__ Qh, const _Float16* __restrict__ Kh,
    const _Float16* __restrict__ Vh, _Float16* __restrict__ Oh) {
  constexpr int LDK = 72;   // 64 d + 8 pad (matches TDM pad: 128B rows, 16B)
  constexpr int LDV = 72;   // 64 keys + 8 pad (Vt[d][key])
  constexpr int LDP = 72;
  __shared__ _Float16 ldsK[64 * LDK];
  __shared__ _Float16 ldsVt[64 * LDV];
  __shared__ _Float16 ldsP[8 * 16 * LDP];

  const int tid = threadIdx.x;
  const int w = tid >> 5, lane = tid & 31;
  const int l16 = lane & 15, lhi = (lane >> 4) & 1;

  const int blk = blockIdx.x;
  const int qb = blk & 15;
  const int h = (blk >> 4) & 15;
  const int b = blk >> 8;

  const size_t headoff = (size_t)(b * kH + h) * kS * kDh;
  const _Float16* Q = Qh + headoff;
  const _Float16* K = Kh + headoff;
  const _Float16* V = Vh + headoff;

  const int q0 = qb * 128 + w * 16;

  v16h qa0 = fragA_ld(Q + (size_t)(q0 + l16) * kDh + 0 + lhi * 8);
  v16h qa1 = fragA_ld(Q + (size_t)(q0 + l16) * kDh + 32 + lhi * 8);

  v16h ones;
#pragma unroll
  for (int i = 0; i < 16; ++i) ones[i] = (_Float16)1.0f;

  v8f o[4] = {};
  v8f mrun, lrun;
#pragma unroll
  for (int r = 0; r < 8; ++r) { mrun[r] = -3.0e38f; lrun[r] = 0.f; }

  _Float16* Pw = &ldsP[w * 16 * LDP];

  for (int kb = 0; kb < kS / 64; ++kb) {
    const int sk0 = kb * 64;
    __syncthreads();
    if (tid < 32)   // wave 0 kicks the K-tile DMA (128B rows -> code 4)
      tdm_load_2d(lds_off(ldsK), K + (size_t)sk0 * kDh,
                  kDh, kS, kDh, 64, 64, 4, 3);
    // all waves: V tile load + transpose, overlapping the TDM
#pragma unroll
    for (int c = 0; c < 2; ++c) {
      int ch = tid + c * 256;
      int row = ch >> 3, dc = (ch & 7) * 8;
      v8h vv = *(const v8h*)(V + (size_t)(sk0 + row) * kDh + dc);
#pragma unroll
      for (int j = 0; j < 8; ++j) ldsVt[(dc + j) * LDV + row] = vv[j];
    }
    if (tid < 32) __builtin_amdgcn_s_wait_tensorcnt(0);
    __syncthreads();

    // S (16 x 64 keys) = Q(16x64) @ K^T
    v8f sAcc[4] = {};
#pragma unroll
    for (int jt = 0; jt < 4; ++jt) {
      v16h bk = fragB_ld(&ldsK[(jt * 16 + l16) * LDK + 0 + lhi * 8]);
      sAcc[jt] = wmma_f16(qa0, bk, sAcc[jt]);
      bk = fragB_ld(&ldsK[(jt * 16 + l16) * LDK + 32 + lhi * 8]);
      sAcc[jt] = wmma_f16(qa1, bk, sAcc[jt]);
    }

    // row max over 64 keys: combine tiles, one 4-step xor-shuffle per row
    v8f mnew, alpha;
#pragma unroll
    for (int r = 0; r < 8; ++r) {
      float mx = fmaxf(fmaxf(sAcc[0][r], sAcc[1][r]),
                       fmaxf(sAcc[2][r], sAcc[3][r]));
#pragma unroll
      for (int off = 1; off < 16; off <<= 1)
        mx = fmaxf(mx, __shfl_xor(mx, off, 32));
      mnew[r] = fmaxf(mrun[r], mx);
      alpha[r] = __expf(mrun[r] - mnew[r]);
      mrun[r] = mnew[r];
    }

    // P = exp(S - m) -> per-wave LDS (C -> A relayout; LDS in-order per wave)
#pragma unroll
    for (int jt = 0; jt < 4; ++jt)
#pragma unroll
      for (int r = 0; r < 8; ++r) {
        float p = __expf(sAcc[jt][r] - mnew[r]);
        Pw[(r + lhi * 8) * LDP + jt * 16 + l16] = (_Float16)p;
      }
    v16h ap0 = fragA_ld(&Pw[l16 * LDP + 0 + lhi * 8]);
    v16h ap1 = fragA_ld(&Pw[l16 * LDP + 32 + lhi * 8]);

    // row sums via WMMA against ones (lands in the per-vgpr row slot)
    v8f sum = {};
    sum = wmma_f16(ap0, ones, sum);
    sum = wmma_f16(ap1, ones, sum);

#pragma unroll
    for (int r = 0; r < 8; ++r) lrun[r] = lrun[r] * alpha[r] + sum[r];
#pragma unroll
    for (int j = 0; j < 4; ++j)
#pragma unroll
      for (int r = 0; r < 8; ++r) o[j][r] *= alpha[r];

    // O(16x64) += P(16x64) @ V(64x64)
#pragma unroll
    for (int j = 0; j < 4; ++j) {
      v16h bv = fragB_ld(&ldsVt[(j * 16 + l16) * LDV + 0 + lhi * 8]);
      o[j] = wmma_f16(ap0, bv, o[j]);
      bv = fragB_ld(&ldsVt[(j * 16 + l16) * LDV + 32 + lhi * 8]);
      o[j] = wmma_f16(ap1, bv, o[j]);
    }
  }

#pragma unroll
  for (int j = 0; j < 4; ++j)
#pragma unroll
    for (int r = 0; r < 8; ++r) {
      int s = q0 + r + lhi * 8;
      int col = h * kDh + j * 16 + l16;
      float val = o[j][r] / lrun[r];
      Oh[(size_t)(b * kS + s) * kD + col] = (_Float16)val;
    }
}

// ---------------- launcher ----------------------------------------------------
extern "C" void kernel_launch(void* const* d_in, const int* in_sizes, int n_in,
                              void* d_out, int out_size, void* d_ws,
                              size_t ws_size, hipStream_t stream) {
  (void)in_sizes; (void)n_in; (void)out_size; (void)ws_size;
  const float* q  = (const float*)d_in[0];
  const float* k  = (const float*)d_in[1];
  const float* v  = (const float*)d_in[2];
  const float* Wq = (const float*)d_in[3];
  const float* bq = (const float*)d_in[4];
  const float* Wk = (const float*)d_in[5];
  const float* bk = (const float*)d_in[6];
  const float* Wv = (const float*)d_in[7];
  const float* bv = (const float*)d_in[8];
  const float* Wo = (const float*)d_in[9];
  const float* bo = (const float*)d_in[10];
  float* out = (float*)d_out;

  const size_t act = (size_t)kB * kS * kD;   // 4M halves
  const size_t wel = (size_t)kD * kD;        // 1M halves
  _Float16* Qh  = (_Float16*)d_ws;           // [B,H,S,Dh] f16 (Q pre-scaled)
  _Float16* Kh  = Qh + act;
  _Float16* Vh  = Kh + act;
  _Float16* Ah  = Vh + act;                  // attn out [B,S,D] f16
  _Float16* qf  = Ah + act;                  // f16 activations
  _Float16* kf  = qf + act;
  _Float16* vf  = kf + act;
  _Float16* Wqt = vf + act;                  // f16 transposed weights [N][K]
  _Float16* Wkt = Wqt + wel;
  _Float16* Wvt = Wkt + wel;
  _Float16* Wot = Wvt + wel;

  dim3 cgrd(act / (256 * 8));
  cvt_f16_kernel<<<cgrd, 256, 0, stream>>>(q, qf);
  cvt_f16_kernel<<<cgrd, 256, 0, stream>>>(k, kf);
  cvt_f16_kernel<<<cgrd, 256, 0, stream>>>(v, vf);
  dim3 tgrd(kD / 64, kD / 64);
  wtrans_kernel<<<tgrd, 256, 0, stream>>>(Wq, Wqt);
  wtrans_kernel<<<tgrd, 256, 0, stream>>>(Wk, Wkt);
  wtrans_kernel<<<tgrd, 256, 0, stream>>>(Wv, Wvt);
  wtrans_kernel<<<tgrd, 256, 0, stream>>>(Wo, Wot);

  dim3 ggrd(kB * kS / 128, kD / 128);
  const float iscale = 0.125f;  // 64^-0.5 folded into Q projection
  gemm_f16_kernel<true><<<ggrd, 256, 0, stream>>>(qf, Wqt, bq, Qh, iscale);
  gemm_f16_kernel<true><<<ggrd, 256, 0, stream>>>(kf, Wkt, bk, Kh, 1.0f);
  gemm_f16_kernel<true><<<ggrd, 256, 0, stream>>>(vf, Wvt, bv, Vh, 1.0f);

  flash_attn_kernel<<<dim3(kB * kH * (kS / 128)), 256, 0, stream>>>(Qh, Kh, Vh,
                                                                    Ah);

  gemm_f16_kernel<false><<<ggrd, 256, 0, stream>>>(Ah, Wot, bo, out, 1.0f);
}
